// TransAdapt_60172491817912
// MI455X (gfx1250) — matrix operand order repaired and checked by reading, not compiled, not run on hardware
//
#include <hip/hip_runtime.h>

// ---------------------------------------------------------------------------
// CDNA5 (gfx1250, wave32) implementation of the deformable-attention block.
// All dense GEMMs use v_wmma_f32_16x16x32_bf16 (bf16 inputs, f32 accum).
// GEMM: block tile 64x128, each wave owns a 16x64 strip -> 4 WMMAs per
// A-fragment (A reuse), 32 WMMAs per staged LDS tile.
// The value tensor v is stored bf16 (134 MB) so bilinear gathers hit the
// 192 MB L2 instead of HBM.
// ---------------------------------------------------------------------------

typedef __bf16 bf16;
typedef __attribute__((ext_vector_type(16))) __bf16 v16bf;
typedef __attribute__((ext_vector_type(8)))  float  v8f;

#define DMODEL 256
#define NHEAD  8
#define PPTS   25
#define HDIM   32
#define DFFN   1024
#define NQ     8192
#define HH     512
#define WWID   512
#define HWSZ   (HH * WWID)

union BF16x16 { v16bf v; unsigned u[8]; };

// ---------------------------------------------------------------------------
// Generic GEMM: out[m][n] = act( sum_k A[m][k] * W[n][k] + bias[n] (+resid) )
// A: (M,Kd) f32 row-major.  W: (Nc,Kd) f32 row-major (i.e. x @ W^T).
// Block: 256 threads = 8 waves (4 M-waves x 2 N-waves).
// Block tile: 64(M) x 128(N), K-step 32. Wave tile: 16(M) x 64(N).
// M, Kd must be multiples of 64/32; Nc is bounds-checked (200, 40 cases).
// ---------------------------------------------------------------------------
__launch_bounds__(256)
__global__ void gemm_bf16_wmma(const float* __restrict__ A,
                               const float* __restrict__ W,
                               const float* __restrict__ bias,
                               const float* __restrict__ resid,
                               float* __restrict__ outF,
                               bf16*  __restrict__ outB,
                               int M, int Nc, int Kd, int relu)
{
    __shared__ __align__(16) bf16 sA[64][36];    // +4 pad, 72B pitch (4B mult)
    __shared__ __align__(16) bf16 sB[128][36];

    const int t    = threadIdx.x;
    const int lane = t & 31;
    const int wave = t >> 5;       // 0..7
    const int wm   = wave & 3;     // 0..3  (M sub-tile)
    const int wn   = wave >> 2;    // 0..1  (N strip of 64)
    const int m0   = blockIdx.x * 64;
    const int n0   = blockIdx.y * 128;

    const int arow = t >> 2;            // 0..63
    const int acol = (t & 3) * 8;       // 0,8,16,24
    const int brow = t >> 1;            // 0..127
    const int bcol = (t & 1) * 16;      // 0,16

    v8f acc0 = {}, acc1 = {}, acc2 = {}, acc3 = {};

    for (int k0 = 0; k0 < Kd; k0 += 32) {
        // ---- stage A tile 64x32 (f32 -> bf16), 8 elems/thread ----
        {
            const float* ag = A + (size_t)(m0 + arow) * Kd + k0 + acol;
            const float4 v0 = *reinterpret_cast<const float4*>(ag);
            const float4 v1 = *reinterpret_cast<const float4*>(ag + 4);
            sA[arow][acol + 0] = (bf16)v0.x;
            sA[arow][acol + 1] = (bf16)v0.y;
            sA[arow][acol + 2] = (bf16)v0.z;
            sA[arow][acol + 3] = (bf16)v0.w;
            sA[arow][acol + 4] = (bf16)v1.x;
            sA[arow][acol + 5] = (bf16)v1.y;
            sA[arow][acol + 6] = (bf16)v1.z;
            sA[arow][acol + 7] = (bf16)v1.w;
        }
        // ---- stage B tile 128x32 = weight rows, 16 elems/thread ----
        {
            const int n = n0 + brow;
            if (n < Nc) {
                const float* wg = W + (size_t)n * Kd + k0 + bcol;
                #pragma unroll
                for (int q = 0; q < 4; ++q) {
                    const float4 vv = *reinterpret_cast<const float4*>(wg + q * 4);
                    sB[brow][bcol + q * 4 + 0] = (bf16)vv.x;
                    sB[brow][bcol + q * 4 + 1] = (bf16)vv.y;
                    sB[brow][bcol + q * 4 + 2] = (bf16)vv.z;
                    sB[brow][bcol + q * 4 + 3] = (bf16)vv.w;
                }
            } else {
                #pragma unroll
                for (int i = 0; i < 16; ++i) sB[brow][bcol + i] = (bf16)0.0f;
            }
        }
        __syncthreads();

        // ---- build fragments, contiguous k order for A and B (wave32, 16-bit A 16x32) ----
        // A: lanes 0-15 hold row M=lane, K=0..15; lanes 16-31 K=16..31
        // B: lanes 0-15 hold col N=lane, K=0..15; lanes 16-31 K=16..31
        BF16x16 afrag;
        {
            const unsigned* ap = reinterpret_cast<const unsigned*>(
                &sA[wm * 16 + (lane & 15)][0]);
            const int ao = (lane < 16) ? 0 : 8;   // uint units (2 bf16 each)
            #pragma unroll
            for (int i = 0; i < 4; ++i) {
                afrag.u[i]     = ap[ao + i];       // K = o .. o+7
                afrag.u[4 + i] = ap[ao + 4 + i];   // K = o+8 .. o+15
            }
        }
        const int bo = (lane < 16) ? 0 : 8;

        BF16x16 bf0, bf1, bf2, bf3;
        {
            const unsigned* bp0 = reinterpret_cast<const unsigned*>(
                &sB[wn * 64 +  0 + (lane & 15)][0]);
            const unsigned* bp1 = reinterpret_cast<const unsigned*>(
                &sB[wn * 64 + 16 + (lane & 15)][0]);
            const unsigned* bp2 = reinterpret_cast<const unsigned*>(
                &sB[wn * 64 + 32 + (lane & 15)][0]);
            const unsigned* bp3 = reinterpret_cast<const unsigned*>(
                &sB[wn * 64 + 48 + (lane & 15)][0]);
            #pragma unroll
            for (int i = 0; i < 8; ++i) {
                bf0.u[i] = bp0[bo + i];
                bf1.u[i] = bp1[bo + i];
                bf2.u[i] = bp2[bo + i];
                bf3.u[i] = bp3[bo + i];
            }
        }

        acc0 = __builtin_amdgcn_wmma_f32_16x16x32_bf16(
            false, afrag.v, false, bf0.v, (short)0, acc0, false, false);
        acc1 = __builtin_amdgcn_wmma_f32_16x16x32_bf16(
            false, afrag.v, false, bf1.v, (short)0, acc1, false, false);
        acc2 = __builtin_amdgcn_wmma_f32_16x16x32_bf16(
            false, afrag.v, false, bf2.v, (short)0, acc2, false, false);
        acc3 = __builtin_amdgcn_wmma_f32_16x16x32_bf16(
            false, afrag.v, false, bf3.v, (short)0, acc3, false, false);

        __syncthreads();
    }

    // ---- epilogue: C/D layout — VGPR g: lanes 0-15 -> M=g, lanes 16-31 -> M=g+8
    const int mb = m0 + wm * 16 + ((lane < 16) ? 0 : 8);
    #pragma unroll
    for (int j = 0; j < 4; ++j) {
        const v8f acc = (j == 0) ? acc0 : (j == 1) ? acc1 : (j == 2) ? acc2 : acc3;
        const int n = n0 + wn * 64 + j * 16 + (lane & 15);
        if (n < Nc) {
            const float bval = bias ? bias[n] : 0.0f;
            #pragma unroll
            for (int g = 0; g < 8; ++g) {
                const size_t o = (size_t)(mb + g) * (size_t)Nc + (size_t)n;
                float v = acc[g] + bval;
                if (resid) v += resid[o];
                if (relu)  v = fmaxf(v, 0.0f);
                if (outF)  outF[o] = v;
                if (outB)  outB[o] = (bf16)v;
            }
        }
    }
}

// ---------------------------------------------------------------------------
// qp1 = relu(ref_windows @ Wp1^T + bp1), K=8 (too small for WMMA)
// ---------------------------------------------------------------------------
__global__ void qp1_kernel(const float* __restrict__ rw,
                           const float* __restrict__ Wp1,
                           const float* __restrict__ bp1,
                           float* __restrict__ out)
{
    int t = blockIdx.x * blockDim.x + threadIdx.x;
    if (t >= NQ * DMODEL) return;
    int n = t >> 8, d = t & 255;
    const float* r = rw  + n * 8;
    const float* w = Wp1 + d * 8;
    float a = bp1[d];
    #pragma unroll
    for (int k = 0; k < 8; ++k) a += r[k] * w[k];
    out[t] = fmaxf(a, 0.0f);
}

// ---------------------------------------------------------------------------
// softmax over P=25 per (n, head); logits layout (n, h*25+p), in-place
// ---------------------------------------------------------------------------
__global__ void softmax25_kernel(float* __restrict__ attn)
{
    int r = blockIdx.x * blockDim.x + threadIdx.x;
    if (r >= NQ * NHEAD) return;
    float* p = attn + (size_t)(r >> 3) * (NHEAD * PPTS) + (r & 7) * PPTS;
    float m = -3.4e38f;
    #pragma unroll
    for (int i = 0; i < PPTS; ++i) m = fmaxf(m, p[i]);
    float s = 0.0f;
    #pragma unroll
    for (int i = 0; i < PPTS; ++i) { float e = __expf(p[i] - m); p[i] = e; s += e; }
    float inv = 1.0f / s;
    #pragma unroll
    for (int i = 0; i < PPTS; ++i) p[i] *= inv;
}

// ---------------------------------------------------------------------------
// sampling locations: per (n, h, p) -> (x, y) in [0,1] units
// ---------------------------------------------------------------------------
__global__ void loc_kernel(const float* __restrict__ rw,
                           const float* __restrict__ off,
                           float* __restrict__ loc)
{
    int t = blockIdx.x * blockDim.x + threadIdx.x;
    if (t >= NQ * NHEAD * PPTS) return;
    int p = t % PPTS;
    int h = (t / PPTS) & (NHEAD - 1);
    int n = t / (NHEAD * PPTS);

    const float* r = rw  + n * 8;
    const float* o = off + n * (NHEAD * 5) + h * 5;

    float cx0 = r[0], cy0 = r[1], w0 = r[3], h0 = r[4], a0 = r[6];
    float angle = (a0 + o[4] * (1.0f / 16.0f)) * 6.28318530717958647692f;
    float cx = cx0 + o[0] * 0.125f * w0;
    float cy = cy0 + o[1] * 0.125f * h0;
    float bw = fmaxf(w0 + o[2] * 0.125f * w0, 0.0f);
    float bh = fmaxf(h0 + o[3] * 0.125f * h0, 0.0f);

    int gi = p / 5, gj = p % 5;                 // kernel grid: ((j-2)/5, (i-2)/5)
    float gx = (float)(gj - 2) * 0.2f * bw;
    float gy = (float)(gi - 2) * 0.2f * bh;
    float c = __cosf(angle), s = __sinf(angle);

    loc[(size_t)t * 2 + 0] = cx + gx * c - gy * s;
    loc[(size_t)t * 2 + 1] = cy + gx * s + gy * c;
}

// ---------------------------------------------------------------------------
// bilinear gather + attention-weighted sum. One wave per (n, head);
// lane = channel. Each corner fetch = 32 contiguous bf16 (64 B) from v,
// which fits in L2 (134 MB < 192 MB).
// ---------------------------------------------------------------------------
__launch_bounds__(256)
__global__ void gather_kernel(const bf16*  __restrict__ v,
                              const float* __restrict__ attn,
                              const float* __restrict__ loc,
                              float* __restrict__ samp)
{
    const int lane = threadIdx.x & 31;
    const int wid  = blockIdx.x * (blockDim.x >> 5) + (threadIdx.x >> 5);
    const int n = wid >> 3;
    const int h = wid & 7;

    const float* at = attn + (size_t)n * (NHEAD * PPTS) + h * PPTS;
    const float* lp = loc  + ((size_t)(n * NHEAD + h) * PPTS) * 2;

    float acc = 0.0f;
    for (int p = 0; p < PPTS; ++p) {
        float a  = at[p];
        float x  = lp[2 * p + 0] * (float)WWID - 0.5f;
        float y  = lp[2 * p + 1] * (float)HH   - 0.5f;
        float x0 = floorf(x), y0 = floorf(y);
        #pragma unroll
        for (int corner = 0; corner < 4; ++corner) {
            float xi = x0 + (float)(corner & 1);
            float yi = y0 + (float)(corner >> 1);
            float w  = (1.0f - fabsf(x - xi)) * (1.0f - fabsf(y - yi));
            bool valid = (xi >= 0.0f) && (xi <= (float)(WWID - 1)) &&
                         (yi >= 0.0f) && (yi <= (float)(HH - 1));
            int cxi = min(max((int)xi, 0), WWID - 1);
            int cyi = min(max((int)yi, 0), HH - 1);
            const bf16* vp = v + ((size_t)(cyi * WWID + cxi)) * DMODEL
                               + h * HDIM + lane;
            __builtin_prefetch(vp, 0, 1);           // global_prefetch_b8
            float g = (float)vp[0];
            acc += (valid ? w : 0.0f) * a * g;
        }
    }
    samp[(size_t)n * DMODEL + h * HDIM + lane] = acc;
}

// ---------------------------------------------------------------------------
// LayerNorm over D=256; one block per row, one element per thread
// ---------------------------------------------------------------------------
__launch_bounds__(256)
__global__ void ln_kernel(const float* __restrict__ x,
                          const float* __restrict__ g,
                          const float* __restrict__ be,
                          float* __restrict__ out)
{
    __shared__ float red[256];
    const int n = blockIdx.x, d = threadIdx.x;
    const float v = x[(size_t)n * DMODEL + d];

    red[d] = v; __syncthreads();
    for (int s = 128; s > 0; s >>= 1) {
        if (d < s) red[d] += red[d + s];
        __syncthreads();
    }
    const float mean = red[0] * (1.0f / DMODEL);
    __syncthreads();

    const float c = v - mean;
    red[d] = c * c; __syncthreads();
    for (int s = 128; s > 0; s >>= 1) {
        if (d < s) red[d] += red[d + s];
        __syncthreads();
    }
    const float var = red[0] * (1.0f / DMODEL);

    out[(size_t)n * DMODEL + d] = c * rsqrtf(var + 1e-5f) * g[d] + be[d];
}

// ---------------------------------------------------------------------------
// host-side launch
// ---------------------------------------------------------------------------
static inline char* ws_take(char*& p, size_t bytes)
{
    char* r = p;
    p += (bytes + 255) & ~(size_t)255;
    return r;
}

extern "C" void kernel_launch(void* const* d_in, const int* in_sizes, int n_in,
                              void* d_out, int out_size, void* d_ws, size_t ws_size,
                              hipStream_t stream)
{
    (void)in_sizes; (void)n_in; (void)out_size; (void)ws_size;

    const float* query  = (const float*)d_in[0];
    const float* memory = (const float*)d_in[1];
    const float* refw   = (const float*)d_in[2];
    // d_in[3] memory_shape, d_in[4] memory_start_idx: constants here
    const float* Wp1 = (const float*)d_in[5];
    const float* bp1 = (const float*)d_in[6];
    const float* Wp2 = (const float*)d_in[7];
    const float* bp2 = (const float*)d_in[8];
    const float* Wp3 = (const float*)d_in[9];
    const float* bp3 = (const float*)d_in[10];
    const float* Wv  = (const float*)d_in[11];
    const float* bv  = (const float*)d_in[12];
    const float* Wa  = (const float*)d_in[13];
    const float* ba  = (const float*)d_in[14];
    const float* Wb  = (const float*)d_in[15];
    const float* bb  = (const float*)d_in[16];
    const float* Wo  = (const float*)d_in[17];
    const float* bo  = (const float*)d_in[18];
    const float* W1  = (const float*)d_in[19];
    const float* b1  = (const float*)d_in[20];
    const float* W2  = (const float*)d_in[21];
    const float* b2  = (const float*)d_in[22];
    const float* g2  = (const float*)d_in[23];
    const float* be2 = (const float*)d_in[24];
    const float* g3  = (const float*)d_in[25];
    const float* be3 = (const float*)d_in[26];

    char* p = (char*)d_ws;
    float* qp1  = (float*)ws_take(p, (size_t)NQ * DMODEL * 4);
    float* qp2  = (float*)ws_take(p, (size_t)NQ * DMODEL * 4);
    float* q    = (float*)ws_take(p, (size_t)NQ * DMODEL * 4);
    bf16*  vbf  = (bf16*) ws_take(p, (size_t)HWSZ * DMODEL * 2);   // 134 MB, fits L2
    float* attn = (float*)ws_take(p, (size_t)NQ * NHEAD * PPTS * 4);
    float* off  = (float*)ws_take(p, (size_t)NQ * NHEAD * 5 * 4);
    float* loc  = (float*)ws_take(p, (size_t)NQ * NHEAD * PPTS * 2 * 4);
    float* samp = (float*)ws_take(p, (size_t)NQ * DMODEL * 4);
    float* pre1 = (float*)ws_take(p, (size_t)NQ * DMODEL * 4);
    float* xq   = (float*)ws_take(p, (size_t)NQ * DMODEL * 4);
    float* h1   = (float*)ws_take(p, (size_t)NQ * DFFN * 4);
    float* pre2 = (float*)ws_take(p, (size_t)NQ * DMODEL * 4);

    const dim3 blk(256);

    // 1. qp1 = relu(refw @ Wp1^T + bp1)
    qp1_kernel<<<dim3((NQ * DMODEL + 255) / 256), blk, 0, stream>>>(refw, Wp1, bp1, qp1);

    // 2. qp2 = relu(qp1 @ Wp2^T + bp2)
    gemm_bf16_wmma<<<dim3(NQ / 64, (DMODEL + 127) / 128), blk, 0, stream>>>(
        qp1, Wp2, bp2, nullptr, qp2, nullptr, NQ, DMODEL, DMODEL, 1);

    // 3. q = query + qp2 @ Wp3^T + bp3
    gemm_bf16_wmma<<<dim3(NQ / 64, (DMODEL + 127) / 128), blk, 0, stream>>>(
        qp2, Wp3, bp3, query, q, nullptr, NQ, DMODEL, DMODEL, 0);

    // 4. v = memory @ Wv^T + bv  (bf16 output -> L2-resident gather source)
    gemm_bf16_wmma<<<dim3(HWSZ / 64, (DMODEL + 127) / 128), blk, 0, stream>>>(
        memory, Wv, bv, nullptr, nullptr, vbf, HWSZ, DMODEL, DMODEL, 0);

    // 5. attn logits = q @ Wa^T + ba   (Nc = 200)
    gemm_bf16_wmma<<<dim3(NQ / 64, (NHEAD * PPTS + 127) / 128), blk, 0, stream>>>(
        q, Wa, ba, nullptr, attn, nullptr, NQ, NHEAD * PPTS, DMODEL, 0);

    // 6. off = q @ Wb^T + bb          (Nc = 40)
    gemm_bf16_wmma<<<dim3(NQ / 64, (NHEAD * 5 + 127) / 128), blk, 0, stream>>>(
        q, Wb, bb, nullptr, off, nullptr, NQ, NHEAD * 5, DMODEL, 0);

    // 7. softmax over 25 points per head
    softmax25_kernel<<<dim3((NQ * NHEAD + 255) / 256), blk, 0, stream>>>(attn);

    // 8. rotated-grid sampling locations
    loc_kernel<<<dim3((NQ * NHEAD * PPTS + 255) / 256), blk, 0, stream>>>(refw, off, loc);

    // 9. bilinear gather + attention sum  (one wave per (n,h))
    gather_kernel<<<dim3(NQ * NHEAD / 8), blk, 0, stream>>>(vbf, attn, loc, samp);

    // 10. pre1 = query + samp @ Wo^T + bo
    gemm_bf16_wmma<<<dim3(NQ / 64, (DMODEL + 127) / 128), blk, 0, stream>>>(
        samp, Wo, bo, query, pre1, nullptr, NQ, DMODEL, DMODEL, 0);

    // 11. xq = LN(pre1)
    ln_kernel<<<dim3(NQ), blk, 0, stream>>>(pre1, g2, be2, xq);

    // 12. h1 = relu(xq @ W1^T + b1)   (Nc = 1024)
    gemm_bf16_wmma<<<dim3(NQ / 64, DFFN / 128), blk, 0, stream>>>(
        xq, W1, b1, nullptr, h1, nullptr, NQ, DFFN, DMODEL, 1);

    // 13. pre2 = xq + h1 @ W2^T + b2  (Kd = 1024)
    gemm_bf16_wmma<<<dim3(NQ / 64, (DMODEL + 127) / 128), blk, 0, stream>>>(
        h1, W2, b2, xq, pre2, nullptr, NQ, DMODEL, DFFN, 0);

    // 14. out = LN(pre2)
    ln_kernel<<<dim3(NQ), blk, 0, stream>>>(pre2, g3, be3, (float*)d_out);
}
